// MultiHeadSelfAttention_89885075571012
// MI455X (gfx1250) — matrix operand order repaired and checked by reading, not compiled, not run on hardware
//
#include <hip/hip_runtime.h>
#include <hip/hip_bf16.h>

// Problem constants (match reference)
#define BB 4
#define SS 2048
#define DDIM 1024
#define HH 16
#define HD 64

#define LPAD 40   // LDS row stride in ushorts (80B: 16B-aligned rows for b128 async stores)

typedef __attribute__((ext_vector_type(16))) __bf16 v16bf;
typedef __attribute__((ext_vector_type(8)))  float  v8f;
typedef __attribute__((ext_vector_type(4)))  int    v4i;

// AS-qualified pointer types for the async-to-LDS builtin
typedef __attribute__((address_space(1))) v4i* gptr_v4i;
typedef __attribute__((address_space(3))) v4i* lptr_v4i;

union Frag16 {
    unsigned int u[8];
    v16bf v;
};

__device__ __forceinline__ unsigned short f2bf(float f) {
    unsigned int u = __float_as_uint(f);
    u += 0x7FFFu + ((u >> 16) & 1u);          // round to nearest even
    return (unsigned short)(u >> 16);
}

__device__ __forceinline__ v8f vzero8() {
    v8f r;
#pragma unroll
    for (int i = 0; i < 8; ++i) r[i] = 0.f;
    return r;
}

// ---- CDNA5 async global->LDS copy (ASYNCcnt path), 16 bytes per lane ----
__device__ __forceinline__ void async_cp16(const unsigned short* g, unsigned short* l) {
#if __has_builtin(__builtin_amdgcn_global_load_async_to_lds_b128)
    __builtin_amdgcn_global_load_async_to_lds_b128(
        (gptr_v4i)(g), (lptr_v4i)(l), 0, 0);
#else
    *(uint4*)l = *(const uint4*)g;   // synchronous fallback
#endif
}

__device__ __forceinline__ void wait_async0() {
#if __has_builtin(__builtin_amdgcn_s_wait_asynccnt)
    __builtin_amdgcn_s_wait_asynccnt(0);
#else
    asm volatile("s_wait_asynccnt 0" ::: "memory");
#endif
}

// A-matrix fragment (16x32 bf16), contiguous k layout (same order as the B loaders):
// lane&15 = row M; lanes 0-15 hold K=0..15, lanes 16-31 hold K=16..31; VGPR v -> K=2v,2v+1.
__device__ __forceinline__ Frag16 load_a_frag(const unsigned short* base, int stride, int lane) {
    Frag16 f;
    int row = lane & 15, half = lane >> 4;
    const unsigned short* p = base + row * stride;
#pragma unroll
    for (int v = 0; v < 8; ++v) {
        int k = 2 * v + 16 * half;
        f.u[v] = *reinterpret_cast<const unsigned int*>(p + k);
    }
    return f;
}

// B-matrix fragment (32x16 bf16) from [n][k] storage (B[k][n] = M[n][k]).
// Lanes 0-15 hold K=0..15, lanes 16-31 hold K=16..31; VGPR v -> K=2v,2v+1.
__device__ __forceinline__ Frag16 load_b_frag_nk(const unsigned short* base, int stride, int lane) {
    Frag16 f;
    int n = lane & 15, half = lane >> 4;
    const unsigned short* p = base + n * stride;
#pragma unroll
    for (int v = 0; v < 8; ++v) {
        int k = 2 * v + 16 * half;
        f.u[v] = *reinterpret_cast<const unsigned int*>(p + k);
    }
    return f;
}

// B-matrix fragment from [k][n] storage (V tile): gather along k.
__device__ __forceinline__ Frag16 load_b_frag_kn(const unsigned short* base, int stride, int lane) {
    Frag16 f;
    int n = lane & 15, half = lane >> 4;
#pragma unroll
    for (int v = 0; v < 8; ++v) {
        int k = 2 * v + 16 * half;
        unsigned int lo = base[(k + 0) * stride + n];
        unsigned int hi = base[(k + 1) * stride + n];
        f.u[v] = lo | (hi << 16);
    }
    return f;
}

__device__ __forceinline__ v8f wmma_bf16(const Frag16& a, const Frag16& b, v8f c) {
    return __builtin_amdgcn_wmma_f32_16x16x32_bf16(false, a.v, false, b.v, (short)0, c, false, false);
}

// ---------------- fp32 -> bf16 conversion ----------------
__global__ void cvt_f32_bf16_kernel(const float* __restrict__ src,
                                    unsigned short* __restrict__ dst, int n) {
    int i = blockIdx.x * blockDim.x + threadIdx.x;
    int stride = gridDim.x * blockDim.x;
    for (; i < n; i += stride) dst[i] = f2bf(src[i]);
}

// ---------------- GEMM: out = A @ W^T + bias --------------------
// A:[M,K] bf16 row-major, W:[N,K] bf16 row-major. Block = 8 waves, block tile 256x64.
// Each wave computes 32x64. W k-chunk (64x32) staged in LDS via async copy, double-buffered.
// MODE 0: bf16 store scattered to [B,H,S,HD]; MODE 1: f32 store row-major.
template <int MODE>
__global__ void __launch_bounds__(256) gemm_kernel(
    const unsigned short* __restrict__ A, const unsigned short* __restrict__ W,
    const float* __restrict__ bias, void* __restrict__ dstv) {
    __shared__ unsigned short wt[2][64][LPAD];

    const int K = DDIM;
    int tid = threadIdx.x;
    int lane = tid & 31, wave = tid >> 5;
    int m0 = (blockIdx.y * 8 + wave) * 32;
    int n0 = blockIdx.x * 64;

    // cooperative copy mapping: 256 threads x 16B = 64 rows x 32 k (bf16)
    int crow = tid >> 2;
    int cseg = (tid & 3) * 8;
    const unsigned short* wsrc = W + (size_t)(n0 + crow) * K + cseg;

    v8f acc[2][4];
#pragma unroll
    for (int tm = 0; tm < 2; ++tm)
#pragma unroll
        for (int t = 0; t < 4; ++t) acc[tm][t] = vzero8();

    // prologue: stage chunk 0
    async_cp16(wsrc + 0, &wt[0][crow][cseg]);
    wait_async0();
    __syncthreads();

    for (int kc = 0; kc < K; kc += 32) {
        int buf = (kc >> 5) & 1;
        if (kc + 32 < K) async_cp16(wsrc + kc + 32, &wt[buf ^ 1][crow][cseg]);

        Frag16 af0 = load_a_frag(A + (size_t)m0 * K + kc, K, lane);
        Frag16 af1 = load_a_frag(A + (size_t)(m0 + 16) * K + kc, K, lane);
#pragma unroll
        for (int t = 0; t < 4; ++t) {
            Frag16 bfrag = load_b_frag_nk(&wt[buf][16 * t][0], LPAD, lane);
            acc[0][t] = wmma_bf16(af0, bfrag, acc[0][t]);
            acc[1][t] = wmma_bf16(af1, bfrag, acc[1][t]);
        }

        if (kc + 32 < K) {
            wait_async0();
            __syncthreads();
        }
    }

    int half = lane >> 4, nl = lane & 15;
#pragma unroll
    for (int tm = 0; tm < 2; ++tm) {
#pragma unroll
        for (int t = 0; t < 4; ++t) {
            int n = n0 + 16 * t + nl;
            float bn = bias[n];
#pragma unroll
            for (int r = 0; r < 8; ++r) {
                int m = m0 + 16 * tm + r + 8 * half;
                float val = acc[tm][t][r] + bn;
                if (MODE == 0) {
                    int h = n >> 6, d = n & 63;
                    int b = m >> 11, s = m & (SS - 1);
                    ((unsigned short*)dstv)[(((b * HH + h) * SS) + s) * HD + d] = f2bf(val);
                } else {
                    ((float*)dstv)[(size_t)m * DDIM + n] = val;
                }
            }
        }
    }
}

// ---------------- fused flash attention -------------------
// Block = 8 waves x 16 queries = 128 queries of one (b,h). 32-key K/V tiles staged in
// LDS via async copy (double-buffered, shared by all 8 waves); online softmax per wave.
__global__ void __launch_bounds__(256) attn_kernel(
    const unsigned short* __restrict__ Q, const unsigned short* __restrict__ Km,
    const unsigned short* __restrict__ V, const unsigned char* __restrict__ mask,
    unsigned short* __restrict__ O /* [B*S, D] bf16 row-major */) {
    __shared__ unsigned short kt[2][32][LPAD];
    __shared__ unsigned short vt[2][32][LPAD];
    __shared__ unsigned short pb[8][16][32];

    int tid = threadIdx.x;
    int lane = tid & 31, wave = tid >> 5;
    int bh = blockIdx.y;
    int b = bh >> 4, h = bh & 15;
    int q0 = (blockIdx.x * 8 + wave) * 16;

    const unsigned short* Qb = Q + (size_t)bh * SS * HD;
    const unsigned short* Kb = Km + (size_t)bh * SS * HD;
    const unsigned short* Vb = V + (size_t)bh * SS * HD;
    const unsigned char* mb = mask + b * SS;

    // cooperative copy mapping: 256 threads x 16B = 32 rows x 64 d (bf16) per tile
    int crow = tid >> 3;
    int cseg = (tid & 7) * 8;

    Frag16 aq0 = load_a_frag(Qb + q0 * HD + 0, HD, lane);
    Frag16 aq1 = load_a_frag(Qb + q0 * HD + 32, HD, lane);

    v8f acc[4];
#pragma unroll
    for (int t = 0; t < 4; ++t) acc[t] = vzero8();
    float mrun[8], lrun[8];
#pragma unroll
    for (int r = 0; r < 8; ++r) { mrun[r] = -INFINITY; lrun[r] = 0.f; }

    int half = lane >> 4, nl = lane & 15;
    const float scale = 0.125f;  // 1/sqrt(HD)

    // prologue: stage key-block 0
    async_cp16(Kb + crow * HD + cseg, &kt[0][crow][cseg]);
    async_cp16(Vb + crow * HD + cseg, &vt[0][crow][cseg]);
    wait_async0();
    __syncthreads();

    for (int kb0 = 0; kb0 < SS; kb0 += 32) {
        int buf = (kb0 >> 5) & 1;
        if (kb0 + 32 < SS) {
            async_cp16(Kb + (kb0 + 32 + crow) * HD + cseg, &kt[buf ^ 1][crow][cseg]);
            async_cp16(Vb + (kb0 + 32 + crow) * HD + cseg, &vt[buf ^ 1][crow][cseg]);
        }

        // scores: 16 queries x 32 keys (two C tiles), K-dim = HD = 2x32, from LDS
        v8f c0 = vzero8(), c1 = vzero8();
        {
            Frag16 b00 = load_b_frag_nk(&kt[buf][0][0], LPAD, lane);
            c0 = wmma_bf16(aq0, b00, c0);
            Frag16 b01 = load_b_frag_nk(&kt[buf][0][32], LPAD, lane);
            c0 = wmma_bf16(aq1, b01, c0);
            Frag16 b10 = load_b_frag_nk(&kt[buf][16][0], LPAD, lane);
            c1 = wmma_bf16(aq0, b10, c1);
            Frag16 b11 = load_b_frag_nk(&kt[buf][16][32], LPAD, lane);
            c1 = wmma_bf16(aq1, b11, c1);
        }

        float mb0 = mb[kb0 + nl] ? 0.f : -1e30f;
        float mb1 = mb[kb0 + 16 + nl] ? 0.f : -1e30f;

        float alpha[8];
#pragma unroll
        for (int r = 0; r < 8; ++r) {
            float s0 = c0[r] * scale + mb0;
            float s1 = c1[r] * scale + mb1;
            float rm = fmaxf(s0, s1);
#pragma unroll
            for (int off = 1; off < 16; off <<= 1)      // reduce within 16-lane half (wave32)
                rm = fmaxf(rm, __shfl_xor(rm, off, 32));
            float mn = fmaxf(mrun[r], rm);
            float a = __expf(mrun[r] - mn);
            float p0 = __expf(s0 - mn);
            float p1 = __expf(s1 - mn);
            float rs = p0 + p1;
#pragma unroll
            for (int off = 1; off < 16; off <<= 1)
                rs += __shfl_xor(rs, off, 32);
            lrun[r] = lrun[r] * a + rs;
            mrun[r] = mn;
            alpha[r] = a;
            c0[r] = p0;
            c1[r] = p1;
        }

#pragma unroll
        for (int t = 0; t < 4; ++t)
#pragma unroll
            for (int r = 0; r < 8; ++r) acc[t][r] *= alpha[r];

        // stage P (C layout) into wave-private LDS as bf16, re-read in A layout.
        // DS ops are in-order per wave: a dscnt wait is all that's needed.
#pragma unroll
        for (int r = 0; r < 8; ++r) {
            int m = r + 8 * half;
            pb[wave][m][nl] = f2bf(c0[r]);
            pb[wave][m][nl + 16] = f2bf(c1[r]);
        }
        asm volatile("s_wait_dscnt 0" ::: "memory");
        Frag16 pa = load_a_frag(&pb[wave][0][0], 32, lane);
        asm volatile("" ::: "memory");

        // PV: accumulate 16x64 output, 4 N-tiles, V from LDS
#pragma unroll
        for (int t = 0; t < 4; ++t) {
            Frag16 vf = load_b_frag_kn(&vt[buf][0][16 * t], LPAD, lane);
            acc[t] = wmma_bf16(pa, vf, acc[t]);
        }

        if (kb0 + 32 < SS) {
            wait_async0();
            __syncthreads();
        }
    }

#pragma unroll
    for (int r = 0; r < 8; ++r) lrun[r] = 1.f / lrun[r];
#pragma unroll
    for (int t = 0; t < 4; ++t) {
#pragma unroll
        for (int r = 0; r < 8; ++r) {
            int m = q0 + r + 8 * half;
            int n = h * HD + 16 * t + nl;
            O[((size_t)b * SS + m) * DDIM + n] = f2bf(acc[t][r] * lrun[r]);
        }
    }
}

extern "C" void kernel_launch(void* const* d_in, const int* in_sizes, int n_in,
                              void* d_out, int out_size, void* d_ws, size_t ws_size,
                              hipStream_t stream) {
    const float* x  = (const float*)d_in[0];
    const unsigned char* mask = (const unsigned char*)d_in[1];
    const float* Wq = (const float*)d_in[2];
    const float* bq = (const float*)d_in[3];
    const float* Wk = (const float*)d_in[4];
    const float* bk = (const float*)d_in[5];
    const float* Wv = (const float*)d_in[6];
    const float* bv = (const float*)d_in[7];
    const float* Wo = (const float*)d_in[8];
    const float* bo = (const float*)d_in[9];

    const size_t NX = (size_t)BB * SS * DDIM;   // 8,388,608
    const size_t NW = (size_t)DDIM * DDIM;      // 1,048,576

    char* ws = (char*)d_ws;
    size_t off = 0;
    unsigned short* xb  = (unsigned short*)(ws + off); off += NX * 2;
    unsigned short* wqb = (unsigned short*)(ws + off); off += NW * 2;
    unsigned short* wkb = (unsigned short*)(ws + off); off += NW * 2;
    unsigned short* wvb = (unsigned short*)(ws + off); off += NW * 2;
    unsigned short* wob = (unsigned short*)(ws + off); off += NW * 2;
    unsigned short* qb  = (unsigned short*)(ws + off); off += NX * 2;
    unsigned short* kb  = (unsigned short*)(ws + off); off += NX * 2;
    unsigned short* vb  = (unsigned short*)(ws + off); off += NX * 2;
    unsigned short* ab  = (unsigned short*)(ws + off); off += NX * 2;

    // fp32 -> bf16
    cvt_f32_bf16_kernel<<<2048, 256, 0, stream>>>(x, xb, (int)NX);
    cvt_f32_bf16_kernel<<<1024, 256, 0, stream>>>(Wq, wqb, (int)NW);
    cvt_f32_bf16_kernel<<<1024, 256, 0, stream>>>(Wk, wkb, (int)NW);
    cvt_f32_bf16_kernel<<<1024, 256, 0, stream>>>(Wv, wvb, (int)NW);
    cvt_f32_bf16_kernel<<<1024, 256, 0, stream>>>(Wo, wob, (int)NW);

    // QKV projections (store into [B,H,S,HD] bf16)
    dim3 gg(DDIM / 64, (BB * SS) / 256);   // (16, 32)
    gemm_kernel<0><<<gg, 256, 0, stream>>>(xb, wqb, bq, (void*)qb);
    gemm_kernel<0><<<gg, 256, 0, stream>>>(xb, wkb, bk, (void*)kb);
    gemm_kernel<0><<<gg, 256, 0, stream>>>(xb, wvb, bv, (void*)vb);

    // fused flash attention -> [B*S, D] bf16
    attn_kernel<<<dim3(SS / 128, BB * HH), 256, 0, stream>>>(qb, kb, vb, mask, ab);

    // output projection -> fp32 d_out
    gemm_kernel<1><<<gg, 256, 0, stream>>>(ab, wob, bo, d_out);
}
